// MaxSimRanker_62801011802913
// MI455X (gfx1250) — matrix operand order repaired and checked by reading, not compiled, not run on hardware
//
#include <hip/hip_runtime.h>
#include <hip/hip_bf16.h>

typedef __attribute__((ext_vector_type(16))) __bf16 bf16x16;
typedef __attribute__((ext_vector_type(8)))  __bf16 bf16x8;
typedef __attribute__((ext_vector_type(4)))  __bf16 bf16x4;
typedef __attribute__((ext_vector_type(8)))  float  f32x8;
typedef __attribute__((ext_vector_type(4)))  float  f32x4;

#define QLEN    32
#define HDIM    128
#define DOC_LEN 200
#define NTILES  13          // ceil(200/16)
#define SPITCH  136         // LDS row pitch in bf16 (68 dwords/row -> conflict-free b128 reads)
#define NROWS   208         // padded doc-token rows in LDS

// ds_swizzle group-of-32 XOR mode: offset = xor_mask<<10 | and_mask(0x1F)
template<int MASK>
__device__ __forceinline__ float swz_max(float x) {
  int y = __builtin_amdgcn_ds_swizzle(__builtin_bit_cast(int, x), (MASK << 10) | 0x1F);
  return __builtin_fmaxf(x, __builtin_bit_cast(float, y));
}
template<int MASK>
__device__ __forceinline__ float swz_add(float x) {
  int y = __builtin_amdgcn_ds_swizzle(__builtin_bit_cast(int, x), (MASK << 10) | 0x1F);
  return x + __builtin_bit_cast(float, y);
}

// One block = one (batch b, candidate j) pair. 4 waves:
//   wave = 2*mtile + parity ; mtile picks q-rows [16*mt,16*mt+16), parity picks n-tiles nt%2.
__global__ __launch_bounds__(128)
void maxsim_score_kernel(const float* __restrict__ q,
                         const int*   __restrict__ pids,
                         const float* __restrict__ vecs,
                         const int*   __restrict__ bounds,
                         float* __restrict__ scores_ws,
                         int*   __restrict__ wpid_ws,
                         int npids, int ndocs)
{
  __shared__ __bf16 dsh[NROWS * SPITCH];   // 56,576 B doc tile in bf16
  __shared__ float  maxpart[4][16];
  __shared__ int    dupflag;

  const int j = blockIdx.x;
  const int b = blockIdx.y;
  const int t = threadIdx.x;
  const int lo = bounds[0];

  // ---- validity + dedupe (parallel scan; result uniform across block) ----
  if (t == 0) dupflag = 0;
  __syncthreads();
  const int myp = pids[(size_t)b * npids + j] - lo;
  const bool inrange = (myp >= 0) & (myp < ndocs);
  if (inrange) {
    const int* row = pids + (size_t)b * npids;
    for (int i = t; i < j; i += 128)
      if (row[i] - lo == myp) dupflag = 1;   // racy same-value store: fine
  }
  __syncthreads();
  const int pid = (inrange && !dupflag) ? myp : -1;
  if (t == 0) {
    wpid_ws[(size_t)b * npids + j] = pid;
    if (pid < 0) scores_ws[(size_t)b * npids + j] = -__builtin_inff();
  }
  if (pid < 0) return;

  // ---- stage doc tile: fp32 [200,128] -> bf16 LDS [208, SPITCH] ----
  {
    const float* dbase = vecs + (size_t)pid * (DOC_LEN * HDIM);
    for (int i = t; i < (DOC_LEN * HDIM) / 4; i += 128) {
      int row = i >> 5;                 // 32 float4 chunks per 128-wide row
      int c4  = (i & 31) << 2;
      f32x4 x = *(const f32x4*)(dbase + row * HDIM + c4);
      bf16x4 y;
      y[0] = (__bf16)x[0]; y[1] = (__bf16)x[1];
      y[2] = (__bf16)x[2]; y[3] = (__bf16)x[3];
      *(bf16x4*)&dsh[row * SPITCH + c4] = y;
    }
    for (int i = t; i < 8 * (SPITCH / 4); i += 128) {  // zero pad rows [200,208)
      int row = DOC_LEN + i / (SPITCH / 4);
      int c4  = (i % (SPITCH / 4)) << 2;
      bf16x4 z = {};
      *(bf16x4*)&dsh[row * SPITCH + c4] = z;
    }
  }
  __syncthreads();

  const int lane = t & 31;
  const int wid  = t >> 5;
  const int mt   = wid >> 1;
  const int par  = wid & 1;

  // ---- A operand (16x32 bf16 per k-step): lane holds row M=lane&15,
  //      K = (lane>>4)*8 + {0..7} and +16 ----
  bf16x16 A[4];
  {
    const float* qrow = q + ((size_t)b * QLEN + mt * 16 + (lane & 15)) * HDIM;
    const int kb = (lane >> 4) * 8;
#pragma unroll
    for (int ks = 0; ks < 4; ++ks) {
      const float* p0 = qrow + ks * 32 + kb;
      f32x4 x0 = *(const f32x4*)(p0);
      f32x4 x1 = *(const f32x4*)(p0 + 4);
      f32x4 x2 = *(const f32x4*)(p0 + 16);
      f32x4 x3 = *(const f32x4*)(p0 + 20);
#pragma unroll
      for (int e = 0; e < 4; ++e) {
        A[ks][e]      = (__bf16)x0[e];
        A[ks][4 + e]  = (__bf16)x1[e];
        A[ks][8 + e]  = (__bf16)x2[e];
        A[ks][12 + e] = (__bf16)x3[e];
      }
    }
  }

  float rm[8];
#pragma unroll
  for (int r = 0; r < 8; ++r) rm[r] = -__builtin_inff();

  const int bcol = lane & 15;          // B: lane holds column N=lane&15
  const int koff = (lane >> 4) * 8;    //    K = (lane>>4)*8 + {0..7} and +16

  // One n-tile: 8x ds_load_b128 -> 4 chained WMMAs -> masked max-fold.
  auto tile_body = [&](int nt) {
    const __bf16* bbase = &dsh[(nt * 16 + bcol) * SPITCH + koff];
    bf16x8 blo[4], bhi[4];
#pragma unroll
    for (int ks = 0; ks < 4; ++ks) {
      blo[ks] = *(const bf16x8*)(bbase + ks * 32);
      bhi[ks] = *(const bf16x8*)(bbase + ks * 32 + 16);
    }
    f32x8 c = {};
#pragma unroll
    for (int ks = 0; ks < 4; ++ks) {
      bf16x16 Bv = __builtin_shufflevector(blo[ks], bhi[ks],
                     0,1,2,3,4,5,6,7,8,9,10,11,12,13,14,15);
      c = __builtin_amdgcn_wmma_f32_16x16x32_bf16(
              false, A[ks], false, Bv, (short)0, c, false, false);
    }
    if (nt * 16 + bcol < DOC_LEN) {    // folds away except nt == 12
#pragma unroll
      for (int r = 0; r < 8; ++r) rm[r] = __builtin_fmaxf(rm[r], c[r]);
    }
  };

  // constexpr loop bounds per parity -> full unroll -> cross-tile load/WMMA pipelining
  if (par == 0) {
#pragma unroll
    for (int nt = 0; nt < NTILES; nt += 2) tile_body(nt);
  } else {
#pragma unroll
    for (int nt = 1; nt < NTILES; nt += 2) tile_body(nt);
  }

  // max over the 16 lanes of each half (the d columns this wave saw)
#pragma unroll
  for (int r = 0; r < 8; ++r) {
    float v = rm[r];
    v = swz_max<1>(v); v = swz_max<2>(v); v = swz_max<4>(v); v = swz_max<8>(v);
    rm[r] = v;
  }
  if ((lane & 15) == 0) {
    const int qh = (lane >> 4) * 8;    // lane0 -> q_local r, lane16 -> q_local r+8
#pragma unroll
    for (int r = 0; r < 8; ++r) maxpart[wid][qh + r] = rm[r];
  }
  __syncthreads();

  if (t < 32) {                        // wave 0: combine parities, sum over 32 q rows
    const int m2 = t >> 4, qq = t & 15;
    float v = __builtin_fmaxf(maxpart[2 * m2][qq], maxpart[2 * m2 + 1][qq]);
    v = swz_add<1>(v); v = swz_add<2>(v); v = swz_add<4>(v);
    v = swz_add<8>(v); v = swz_add<16>(v);
    if (t == 0) scores_ws[(size_t)b * npids + j] = v;
  }
}

// Iterative top-k (argmax selection) per batch row: 256 threads over 500 candidates.
__global__ __launch_bounds__(256)
void maxsim_topk_kernel(const float* __restrict__ scores,
                        const int*   __restrict__ wpid,
                        const int*   __restrict__ bounds,
                        float* __restrict__ out_scores,
                        int*   __restrict__ out_pids,
                        int npids, int k)
{
  __shared__ float s[512];
  __shared__ float rs[256];
  __shared__ int   ri[256];
  const int b = blockIdx.x;
  const int t = threadIdx.x;
  const int lo = bounds[0];

  for (int i = t; i < 512; i += 256)
    s[i] = (i < npids) ? scores[(size_t)b * npids + i] : -__builtin_inff();
  __syncthreads();

  for (int it = 0; it < k; ++it) {
    float a0 = s[t], a1 = s[t + 256];
    if (a1 > a0) { rs[t] = a1; ri[t] = t + 256; }
    else         { rs[t] = a0; ri[t] = t; }
    __syncthreads();
    for (int off = 128; off > 0; off >>= 1) {
      if (t < off) {
        float vo = rs[t + off]; int io = ri[t + off];
        if (vo > rs[t] || (vo == rs[t] && io < ri[t])) { rs[t] = vo; ri[t] = io; }
      }
      __syncthreads();
    }
    if (t == 0) {
      const int ji = ri[0];
      out_scores[(size_t)b * k + it] = rs[0];
      const int p = wpid[(size_t)b * npids + ji];
      out_pids[(size_t)b * k + it] = (p >= 0) ? p + lo : p;
      s[ji] = -__builtin_inff();
    }
    __syncthreads();
  }
}

extern "C" void kernel_launch(void* const* d_in, const int* in_sizes, int n_in,
                              void* d_out, int out_size, void* d_ws, size_t ws_size,
                              hipStream_t stream)
{
  const float* qv     = (const float*)d_in[0];
  const int*   pids   = (const int*)  d_in[1];
  const float* vecs   = (const float*)d_in[2];
  const int*   bounds = (const int*)  d_in[3];
  (void)n_in; (void)ws_size;

  const int B     = in_sizes[0] / (QLEN * HDIM);       // 16
  const int npids = in_sizes[1] / B;                   // 500
  const int ndocs = in_sizes[2] / (DOC_LEN * HDIM);    // 10000
  const int k     = out_size / (2 * B);                // 100

  float* scores_ws = (float*)d_ws;                                    // B*npids f32
  int*   wpid_ws   = (int*)((char*)d_ws + sizeof(float) * B * npids); // B*npids i32

  float* out_scores = (float*)d_out;                   // [B, k] f32
  int*   out_pids   = (int*)d_out + (size_t)B * k;     // [B, k] i32 (bit layout)

  dim3 g1(npids, B);
  maxsim_score_kernel<<<g1, 128, 0, stream>>>(qv, pids, vecs, bounds,
                                              scores_ws, wpid_ws, npids, ndocs);
  maxsim_topk_kernel<<<B, 256, 0, stream>>>(scores_ws, wpid_ws, bounds,
                                            out_scores, out_pids, npids, k);
}
